// S_Mamba_Expert_42047729828167
// MI455X (gfx1250) — matrix operand order repaired and checked by reading, not compiled, not run on hardware
//
#include <hip/hip_runtime.h>
#include <hip/hip_bf16.h>

// ---------------------------------------------------------------------------
// Types for WMMA (gfx1250, wave32)
// ---------------------------------------------------------------------------
typedef __attribute__((ext_vector_type(16))) __bf16       v16bf;
typedef __attribute__((ext_vector_type(8)))  float        v8f;
typedef __attribute__((ext_vector_type(4)))  float        fx4;
typedef __attribute__((ext_vector_type(4)))  unsigned int ux4;

union FragBF { v16bf v; ux4 q[2]; };

__device__ __forceinline__ unsigned short f2bf(float f) {
  unsigned u = __float_as_uint(f);
  u += 0x7FFFu + ((u >> 16) & 1u);   // round-to-nearest-even
  return (unsigned short)(u >> 16);
}

// ---------------------------------------------------------------------------
// GEMM: C[M,N] = act( A[M,K] (lda) * B[N,K]^T (ldb) + bias[N] )
// fp32 in/out, bf16 WMMA core. Tile 128x64, 256 threads = 8 waves,
// each wave owns a 32x32 sub-tile (2x2 v_wmma_f32_16x16x32_bf16 frags).
// act: 0=none, 1=exact GELU, 2=softplus
// Requires M%128==0, N%64==0, K%32==0 (true for all calls here).
// ---------------------------------------------------------------------------
__global__ __launch_bounds__(256)
void gemm_wmma_bf16(const float* __restrict__ A, int lda,
                    const float* __restrict__ B, int ldb,
                    float* __restrict__ C, int ldc,
                    int M, int N, int K,
                    const float* __restrict__ bias, int act)
{
  __shared__ alignas(16) unsigned short sA[128 * 32];
  __shared__ alignas(16) unsigned short sB[64 * 32];

  const int tid    = threadIdx.x;
  const int lane   = tid & 31;
  const int wave   = tid >> 5;      // 0..7
  const int wr     = wave & 3;      // wave row group (4 x 32 rows)
  const int wc     = wave >> 2;     // wave col group (2 x 32 cols)
  const int laneM  = lane & 15;
  const int laneHi = lane >> 4;     // 0/1
  const int rowBlk = blockIdx.y * 128;
  const int colBlk = blockIdx.x * 64;

  v8f acc[2][2] = {};

  for (int kt = 0; kt < K; kt += 32) {
    // --- stage A tile 128x32 (fp32 -> bf16), 4 float4 per thread ---
#pragma unroll
    for (int c = 0; c < 4; ++c) {
      int ch = tid + c * 256;                 // 0..1023
      int r  = ch >> 3;
      int c4 = (ch & 7) << 2;
      fx4 f = *(const fx4*)(A + (size_t)(rowBlk + r) * lda + kt + c4);
      int o = r * 32 + c4;
      sA[o + 0] = f2bf(f.x); sA[o + 1] = f2bf(f.y);
      sA[o + 2] = f2bf(f.z); sA[o + 3] = f2bf(f.w);
    }
    // --- stage B tile 64x32, 2 float4 per thread ---
#pragma unroll
    for (int c = 0; c < 2; ++c) {
      int ch = tid + c * 256;                 // 0..511
      int r  = ch >> 3;
      int c4 = (ch & 7) << 2;
      fx4 f = *(const fx4*)(B + (size_t)(colBlk + r) * ldb + kt + c4);
      int o = r * 32 + c4;
      sB[o + 0] = f2bf(f.x); sB[o + 1] = f2bf(f.y);
      sB[o + 2] = f2bf(f.z); sB[o + 3] = f2bf(f.w);
    }
    __syncthreads();

    // --- build fragments per ISA 7.12.2 layouts ---
    FragBF a[2], b[2];
#pragma unroll
    for (int i = 0; i < 2; ++i) {
      // A 16x32: lanes 0-15 row M, K {0..7,16..23}; lanes 16-31 K {8..15,24..31}
      int row = wr * 32 + i * 16 + laneM;
      const unsigned short* base = &sA[row * 32 + laneHi * 8];
      a[i].q[0] = *(const ux4*)(base);
      a[i].q[1] = *(const ux4*)(base + 16);
    }
#pragma unroll
    for (int j = 0; j < 2; ++j) {
      // B 32x16: lane n = lane%16; lanes 0-15 hold K {0..7,16..23}, lanes 16-31 K {8..15,24..31}
      int col = wc * 32 + j * 16 + laneM;
      const unsigned short* base = &sB[col * 32 + laneHi * 8];
      b[j].q[0] = *(const ux4*)(base);
      b[j].q[1] = *(const ux4*)(base + 16);
    }
#pragma unroll
    for (int i = 0; i < 2; ++i)
#pragma unroll
      for (int j = 0; j < 2; ++j)
        acc[i][j] = __builtin_amdgcn_wmma_f32_16x16x32_bf16(
            false, a[i].v, false, b[j].v, (short)0, acc[i][j], false, false);
    __syncthreads();
  }

  // --- epilogue: C/D layout lane<16: M=r, N=lane ; lane>=16: M=8+r, N=lane-16 ---
#pragma unroll
  for (int i = 0; i < 2; ++i)
#pragma unroll
    for (int j = 0; j < 2; ++j) {
      int col = colBlk + wc * 32 + j * 16 + laneM;
      float bv = bias ? bias[col] : 0.0f;
      int rowBase = rowBlk + wr * 32 + i * 16 + laneHi * 8;
#pragma unroll
      for (int r = 0; r < 8; ++r) {
        float v = acc[i][j][r] + bv;
        if (act == 1)      v = 0.5f * v * (1.0f + erff(v * 0.70710678118654752f));
        else if (act == 2) v = (v > 20.0f) ? v : log1pf(__expf(v));
        C[(size_t)(rowBase + r) * ldc + col] = v;
      }
    }
}

// ---------------------------------------------------------------------------
// Causal depthwise conv (K=2) + bias + SiLU.  xi = xz[:, 0:512].
// ---------------------------------------------------------------------------
__global__ __launch_bounds__(256)
void conv_silu_kernel(const float* __restrict__ xz, const float* __restrict__ w,
                      const float* __restrict__ cb, float* __restrict__ xc, int L)
{
  int gid = blockIdx.x * 256 + threadIdx.x;   // over M*512
  int d = gid & 511;
  int m = gid >> 9;
  int l = m % L;
  float cur  = xz[(size_t)m * 1024 + d];
  float prev = (l == 0) ? 0.0f : xz[(size_t)(m - 1) * 1024 + d];
  float v = w[d * 2 + 0] * prev + w[d * 2 + 1] * cur + cb[d];
  xc[gid] = v / (1.0f + __expf(-v));
}

// ---------------------------------------------------------------------------
// Selective scan: one thread per (b,d), N=16 states in registers.
// dA and dt*B*u recomputed on the fly (no [B,L,D,N] materialization).
// ---------------------------------------------------------------------------
__global__ __launch_bounds__(256)
void scan_kernel(const float* __restrict__ dt,    // [B*L,512] post-softplus
                 const float* __restrict__ proj,  // [B*L,64]: cols 32..47=Bm, 48..63=Cm
                 const float* __restrict__ xc,    // [B*L,512]
                 const float* __restrict__ A_log, // [512,16]
                 const float* __restrict__ Dskip, // [512]
                 float* __restrict__ y, int L)
{
  int gid = blockIdx.x * 256 + threadIdx.x;       // B*512 threads
  int d = gid & 511;
  int b = gid >> 9;
  float Ac[16], h[16];
#pragma unroll
  for (int n = 0; n < 16; ++n) { Ac[n] = -__expf(A_log[d * 16 + n]); h[n] = 0.0f; }
  float dsk = Dskip[d];
  for (int l = 0; l < L; ++l) {
    size_t m = (size_t)b * L + l;
    float dtv = dt[m * 512 + d];
    float xcv = xc[m * 512 + d];
    const float* pr = proj + m * 64;
    float accv = 0.0f;
#pragma unroll
    for (int n = 0; n < 16; ++n) {
      float dA = __expf(dtv * Ac[n]);
      h[n] = dA * h[n] + dtv * pr[32 + n] * xcv;
      accv += h[n] * pr[48 + n];
    }
    y[m * 512 + d] = accv + dsk * xcv;
  }
}

// y *= silu(z),  z = xz[:, 512:1024]
__global__ __launch_bounds__(256)
void gate_kernel(float* __restrict__ y, const float* __restrict__ xz)
{
  int gid = blockIdx.x * 256 + threadIdx.x;
  int d = gid & 511;
  int m = gid >> 9;
  float z = xz[(size_t)m * 1024 + 512 + d];
  y[gid] *= z / (1.0f + __expf(-z));
}

__global__ __launch_bounds__(256)
void flip_kernel(const float* __restrict__ in, float* __restrict__ out, int L)
{
  int gid = blockIdx.x * 256 + threadIdx.x;
  int d = gid & 511;
  int m = gid >> 9;
  int b = m / L, l = m % L;
  out[gid] = in[((size_t)b * L + (L - 1 - l)) * 512 + d];
}

// out = x + fwd + flip(rev)
__global__ __launch_bounds__(256)
void combine_kernel(const float* __restrict__ x, const float* __restrict__ fwd,
                    const float* __restrict__ rev, float* __restrict__ out, int L)
{
  int gid = blockIdx.x * 256 + threadIdx.x;
  int d = gid & 511;
  int m = gid >> 9;
  int b = m / L, l = m % L;
  out[gid] = x[gid] + fwd[gid] + rev[((size_t)b * L + (L - 1 - l)) * 512 + d];
}

// LayerNorm over D=512, optional residual add: out = LN(in + res)
__global__ __launch_bounds__(128)
void layernorm_kernel(const float* __restrict__ in, const float* __restrict__ res,
                      const float* __restrict__ w, const float* __restrict__ bb,
                      float* __restrict__ out)
{
  __shared__ float red[8];
  size_t row = blockIdx.x;
  const float* p  = in + row * 512;
  const float* pr = res ? res + row * 512 : nullptr;
  float v[4], s = 0.0f, sq = 0.0f;
#pragma unroll
  for (int i = 0; i < 4; ++i) {
    int c = threadIdx.x + i * 128;
    float x = p[c];
    if (pr) x += pr[c];
    v[i] = x; s += x; sq += x * x;
  }
  for (int o = 16; o > 0; o >>= 1) {
    s  += __shfl_down(s, o, 32);
    sq += __shfl_down(sq, o, 32);
  }
  int wv = threadIdx.x >> 5;
  if ((threadIdx.x & 31) == 0) { red[wv] = s; red[4 + wv] = sq; }
  __syncthreads();
  if (threadIdx.x == 0) {
    float ts = 0, tq = 0;
    for (int i = 0; i < 4; ++i) { ts += red[i]; tq += red[4 + i]; }
    float mean = ts / 512.0f;
    red[0] = mean;
    red[4] = tq / 512.0f - mean * mean;
  }
  __syncthreads();
  float mean = red[0];
  float inv = rsqrtf(red[4] + 1e-5f);
#pragma unroll
  for (int i = 0; i < 4; ++i) {
    int c = threadIdx.x + i * 128;
    out[row * 512 + c] = (v[i] - mean) * inv * w[c] + bb[c];
  }
}

// ---------------------------------------------------------------------------
// Host orchestration
// ---------------------------------------------------------------------------
static inline void run_gemm(const float* A, int lda, const float* B, int ldb,
                            float* C, int ldc, int M, int N, int K,
                            const float* bias, int act, hipStream_t s)
{
  dim3 g(N / 64, M / 128);
  gemm_wmma_bf16<<<g, 256, 0, s>>>(A, lda, B, ldb, C, ldc, M, N, K, bias, act);
}

extern "C" void kernel_launch(void* const* d_in, const int* in_sizes, int n_in,
                              void* d_out, int out_size, void* d_ws, size_t ws_size,
                              hipStream_t stream)
{
  constexpr int Bsz = 16, L = 512, D = 512, M = Bsz * L;   // 8192 tokens
  constexpr size_t MB = 1u << 20;

  const float* x_in    = (const float*)d_in[0];
  const float* in_w    = (const float*)d_in[1];
  const float* conv_w  = (const float*)d_in[2];
  const float* conv_b  = (const float*)d_in[3];
  const float* xproj_w = (const float*)d_in[4];
  const float* dt_w    = (const float*)d_in[5];
  const float* dt_b    = (const float*)d_in[6];
  const float* A_log   = (const float*)d_in[7];
  const float* D_skip  = (const float*)d_in[8];
  const float* out_w   = (const float*)d_in[9];
  const float* ffn1_w  = (const float*)d_in[10];
  const float* ffn1_b  = (const float*)d_in[11];
  const float* ffn2_w  = (const float*)d_in[12];
  const float* ffn2_b  = (const float*)d_in[13];
  const float* n1_w    = (const float*)d_in[14];
  const float* n1_b    = (const float*)d_in[15];
  const float* n2_w    = (const float*)d_in[16];
  const float* n2_b    = (const float*)d_in[17];
  const float* fn_w    = (const float*)d_in[18];
  const float* fn_b    = (const float*)d_in[19];

  char* ws = (char*)d_ws;
  // Layout (146 MB): x | [xz|fwd|rev]=64MB region reused as ffn_h | xc | proj | dt(=ffn_y) | y | xr
  float* buf_x    = (float*)(ws);
  float* buf_big  = (float*)(ws + 16 * MB);
  float* buf_xz   = buf_big;                            // 32 MB
  float* buf_fwd  = (float*)(ws + 48 * MB);             // 16 MB
  float* buf_rev  = (float*)(ws + 64 * MB);             // 16 MB
  float* buf_xc   = (float*)(ws + 80 * MB);
  float* buf_proj = (float*)(ws + 96 * MB);
  float* buf_dt   = (float*)(ws + 98 * MB);             // reused as FFN out
  float* buf_y    = (float*)(ws + 114 * MB);
  float* buf_xr   = (float*)(ws + 130 * MB);

  const int EW = 16384;  // elementwise grid = M*512/256

  hipMemcpyAsync(buf_x, x_in, (size_t)M * D * sizeof(float),
                 hipMemcpyDeviceToDevice, stream);

  for (int e = 0; e < 2; ++e) {
    for (int dir = 0; dir < 2; ++dir) {
      int idx = e * 2 + dir;
      const float* xin = buf_x;
      if (dir == 1) {
        flip_kernel<<<EW, 256, 0, stream>>>(buf_x, buf_xr, L);
        xin = buf_xr;
      }
      // xz = x @ in_w^T                  [8192,1024]
      run_gemm(xin, D, in_w + (size_t)idx * 1024 * 512, D,
               buf_xz, 1024, M, 1024, 512, nullptr, 0, stream);
      // xc = silu(causal_conv(xi)+b)     [8192,512]
      conv_silu_kernel<<<EW, 256, 0, stream>>>(buf_xz, conv_w + (size_t)idx * 1024,
                                               conv_b + (size_t)idx * 512, buf_xc, L);
      // proj = xc @ xproj_w^T            [8192,64]
      run_gemm(buf_xc, D, xproj_w + (size_t)idx * 64 * 512, D,
               buf_proj, 64, M, 64, 512, nullptr, 0, stream);
      // dt = softplus(proj[:, :32] @ dt_w^T + dt_b)   [8192,512]
      run_gemm(buf_proj, 64, dt_w + (size_t)idx * 512 * 32, 32,
               buf_dt, D, M, 512, 32, dt_b + (size_t)idx * 512, 2, stream);
      // selective scan -> y
      scan_kernel<<<(Bsz * D) / 256, 256, 0, stream>>>(
          buf_dt, buf_proj, buf_xc, A_log + (size_t)idx * 512 * 16,
          D_skip + (size_t)idx * 512, buf_y, L);
      // y *= silu(z)
      gate_kernel<<<EW, 256, 0, stream>>>(buf_y, buf_xz);
      // dir_out = y @ out_w^T            [8192,512]
      run_gemm(buf_y, D, out_w + (size_t)idx * 512 * 512, D,
               dir ? buf_rev : buf_fwd, D, M, 512, 512, nullptr, 0, stream);
    }
    // x = LN1(x + fwd + flip(rev))
    combine_kernel<<<EW, 256, 0, stream>>>(buf_x, buf_fwd, buf_rev, buf_y, L);
    layernorm_kernel<<<M, 128, 0, stream>>>(buf_y, nullptr,
                                            n1_w + (size_t)e * 512, n1_b + (size_t)e * 512, buf_x);
    // FFN: h = gelu(x @ W1^T + b1)  [8192,2048] -> y = h @ W2^T + b2 -> x = LN2(x+y)
    run_gemm(buf_x, D, ffn1_w + (size_t)e * 2048 * 512, D,
             buf_big, 2048, M, 2048, 512, ffn1_b + (size_t)e * 2048, 1, stream);
    run_gemm(buf_big, 2048, ffn2_w + (size_t)e * 512 * 2048, 2048,
             buf_dt, D, M, 512, 2048, ffn2_b + (size_t)e * 512, 0, stream);
    layernorm_kernel<<<M, 128, 0, stream>>>(buf_dt, buf_x,
                                            n2_w + (size_t)e * 512, n2_b + (size_t)e * 512, buf_x);
  }
  // final LN -> d_out
  layernorm_kernel<<<M, 128, 0, stream>>>(buf_x, nullptr, fn_w, fn_b, (float*)d_out);
}
